// MLA_16381005267337
// MI455X (gfx1250) — compile-verified, matrix operand order checked by reading, not run on hardware
//
#include <hip/hip_runtime.h>

typedef __bf16 bf16_t;
typedef __attribute__((ext_vector_type(8)))  __bf16 v8bf;
typedef __attribute__((ext_vector_type(16))) __bf16 v16bf;
typedef __attribute__((ext_vector_type(8)))  float  v8f;

#define Bc 2
#define Sc 2048
#define Dc 3072
#define Hc 24
#define DHc 128
#define QPc 1536
#define KVPc 2048
#define KVOc 2112            // KVP + ROPE
#define NKVc 4608            // D + H*NOPE
#define M4c 4096             // B*S
#define NEG_BIG (-1.0e30f)

__device__ __forceinline__ v16bf load16(const bf16_t* p) {
    const v8bf* q = (const v8bf*)p;
    v8bf a = q[0], b = q[1];
    v16bf r;
#pragma unroll
    for (int i = 0; i < 8; ++i) { r[i] = a[i]; r[i + 8] = b[i]; }
    return r;
}

__device__ __forceinline__ v8f wmma_bf16(v16bf a, v16bf b, v8f c) {
    return __builtin_amdgcn_wmma_f32_16x16x32_bf16(false, a, false, b, (short)0, c,
                                                   false, false);
}

// 16B async copy: global memory -> LDS, tracked by ASYNCcnt (CDNA5 TDM-lite path)
__device__ __forceinline__ void async_copy16(unsigned lds_off, const bf16_t* g) {
    asm volatile("global_load_async_to_lds_b128 %0, %1, off"
                 :: "v"(lds_off), "v"(g) : "memory");
}

// ---------------------------------------------------------------- converts
__global__ __launch_bounds__(256) void k_f32_to_bf16(const float* __restrict__ x,
                                                     bf16_t* __restrict__ y, size_t n) {
    size_t i = (size_t)blockIdx.x * 256 + threadIdx.x;
    if (i < n) y[i] = (bf16_t)x[i];
}

// W: R x C  ->  Wt: C x R (bf16)
__global__ __launch_bounds__(256) void k_transpose_bf16(const float* __restrict__ W,
                                                        bf16_t* __restrict__ Wt,
                                                        int R, int C) {
    __shared__ float tile[32][33];
    int rb = blockIdx.y * 32, cb = blockIdx.x * 32;
    int tx = threadIdx.x & 31, ty = threadIdx.x >> 5;   // 32 x 8
#pragma unroll
    for (int rr = ty; rr < 32; rr += 8) {
        int r = rb + rr, c = cb + tx;
        tile[rr][tx] = (r < R && c < C) ? W[(size_t)r * C + c] : 0.f;
    }
    __syncthreads();
#pragma unroll
    for (int rr = ty; rr < 32; rr += 8) {
        int c = cb + rr, r = rb + tx;
        if (c < C && r < R) Wt[(size_t)c * R + r] = (bf16_t)tile[tx][rr];
    }
}

// ---------------------------------------------------------------- layernorm
__global__ __launch_bounds__(256) void k_layernorm(const float* __restrict__ X,
                                                   const float* __restrict__ g,
                                                   const float* __restrict__ bta,
                                                   bf16_t* __restrict__ Y,
                                                   int L, int strideX, int strideY) {
    int row = blockIdx.x, tid = threadIdx.x;
    const float* x = X + (size_t)row * strideX;
    bf16_t* y = Y + (size_t)row * strideY;
    __shared__ float red[256];
    float s = 0.f;
    for (int i = tid; i < L; i += 256) s += x[i];
    red[tid] = s; __syncthreads();
    for (int off = 128; off; off >>= 1) {
        if (tid < off) red[tid] += red[tid + off];
        __syncthreads();
    }
    float mu = red[0] / (float)L;
    __syncthreads();
    float v = 0.f;
    for (int i = tid; i < L; i += 256) { float d = x[i] - mu; v += d * d; }
    red[tid] = v; __syncthreads();
    for (int off = 128; off; off >>= 1) {
        if (tid < off) red[tid] += red[tid + off];
        __syncthreads();
    }
    float inv = rsqrtf(red[0] / (float)L + 1e-5f);
    for (int i = tid; i < L; i += 256)
        y[i] = (bf16_t)((x[i] - mu) * inv * g[i] + bta[i]);
}

// ---------------------------------------------------------------- GEMM (NT, bf16 in, f32 out)
// C[M,N] = A[M,K] * Bt[N,K]^T   (tiles staged with async global->LDS copies)
#define BM 128
#define BN 128
#define BK 64
__global__ __launch_bounds__(256) void k_gemm_nt(const bf16_t* __restrict__ A,
                                                 const bf16_t* __restrict__ Bt,
                                                 float* __restrict__ C,
                                                 int M, int N, int K) {
    __shared__ bf16_t As[BM][BK + 8];
    __shared__ bf16_t Bs[BN][BK + 8];
    int tid = threadIdx.x;
    int wave = tid >> 5, lane = tid & 31;
    int lh = lane >> 4, ll = lane & 15;
    int wm = wave >> 1, wn = wave & 1;          // 4 x 2 wave grid
    int m0 = blockIdx.y * BM, n0 = blockIdx.x * BN;

    v8f acc[2][4];
#pragma unroll
    for (int i = 0; i < 2; ++i)
#pragma unroll
        for (int j = 0; j < 4; ++j) acc[i][j] = (v8f)0.f;

    int lrow = tid >> 1;
    int lcol = (tid & 1) * 32;

    int arow = m0 + lrow; if (arow >= M) arow = M - 1;   // clamp (values unused OOB)
    int brow = n0 + lrow; if (brow >= N) brow = N - 1;
    const bf16_t* aptr = A  + (size_t)arow * K + lcol;
    const bf16_t* bptr = Bt + (size_t)brow * K + lcol;

    unsigned aLds[4], bLds[4];
#pragma unroll
    for (int i = 0; i < 4; ++i) {
        aLds[i] = (unsigned)(uintptr_t)&As[lrow][lcol + 8 * i];
        bLds[i] = (unsigned)(uintptr_t)&Bs[lrow][lcol + 8 * i];
    }

    for (int k0 = 0; k0 < K; k0 += BK) {
        __syncthreads();                         // previous tiles fully consumed
#pragma unroll
        for (int i = 0; i < 4; ++i) async_copy16(aLds[i], aptr + k0 + 8 * i);
#pragma unroll
        for (int i = 0; i < 4; ++i) async_copy16(bLds[i], bptr + k0 + 8 * i);
        if (k0 + BK < K) {                       // warm cache for next stage
            __builtin_prefetch(aptr + k0 + BK, 0, 0);
            __builtin_prefetch(bptr + k0 + BK, 0, 0);
        }
        asm volatile("s_wait_asynccnt 0" ::: "memory");
        __syncthreads();                         // tiles resident in LDS
#pragma unroll
        for (int kk = 0; kk < BK; kk += 32) {
            v16bf af[2], bfv[4];
#pragma unroll
            for (int i = 0; i < 2; ++i)
                af[i] = load16(&As[wm * 32 + i * 16 + ll][kk + lh * 16]);
#pragma unroll
            for (int j = 0; j < 4; ++j)
                bfv[j] = load16(&Bs[wn * 64 + j * 16 + ll][kk + lh * 16]);
#pragma unroll
            for (int i = 0; i < 2; ++i)
#pragma unroll
                for (int j = 0; j < 4; ++j)
                    acc[i][j] = wmma_bf16(af[i], bfv[j], acc[i][j]);
        }
    }
#pragma unroll
    for (int i = 0; i < 2; ++i)
#pragma unroll
        for (int j = 0; j < 4; ++j)
#pragma unroll
            for (int r = 0; r < 8; ++r) {
                int row = m0 + wm * 32 + i * 16 + 8 * lh + r;
                int col = n0 + wn * 64 + j * 16 + ll;
                if (row < M && col < N) C[(size_t)row * N + col] = acc[i][j][r];
            }
}

// ---------------------------------------------------------------- RoPE on Q + pack
__global__ __launch_bounds__(256) void k_rope_q(const float* __restrict__ Q,
                                                bf16_t* __restrict__ Qb) {
    size_t idx = (size_t)blockIdx.x * 256 + threadIdx.x;
    if (idx >= (size_t)M4c * Hc * DHc) return;
    int d = idx & 127;
    size_t t = idx >> 7;
    int h = (int)(t % Hc);
    size_t bs = t / Hc;
    int s = (int)(bs % Sc), b = (int)(bs / Sc);
    const float* q = Q + ((size_t)b * Sc + s) * Dc + h * DHc;
    float val;
    if (d < 64) {
        val = q[d];
    } else {
        int j = d - 64, i = j & 31;
        float freq = powf(10000.f, -(float)i * (1.f / 32.f));
        float sn, cs; sincosf((float)s * freq, &sn, &cs);
        float rot = (j < 32) ? -q[64 + j + 32] : q[64 + j - 32];
        val = q[d] * cs + rot * sn;
    }
    Qb[(((size_t)(b * Hc + h)) * Sc + s) * DHc + d] =
        (bf16_t)(val * 0.08838834764831845f);   // 1/sqrt(128) folded into Q
}

// ---------------------------------------------------------------- build K (nope|rope) and V^T
__global__ __launch_bounds__(256) void k_build_kv(const float* __restrict__ KV,
                                                  const float* __restrict__ CKV,
                                                  bf16_t* __restrict__ Kb,
                                                  bf16_t* __restrict__ Vt) {
    size_t idx = (size_t)blockIdx.x * 256 + threadIdx.x;
    if (idx >= (size_t)M4c * Hc * DHc) return;
    int d = idx & 127;
    size_t t = idx >> 7;
    int h = (int)(t % Hc);
    size_t bs = t / Hc;
    int s = (int)(bs % Sc), b = (int)(bs / Sc);
    size_t row = (size_t)b * Sc + s;
    float kval;
    if (d < 64) {
        kval = KV[row * NKVc + h * 192 + d];
    } else {
        int j = d - 64, i = j & 31;
        float freq = powf(10000.f, -(float)i * (1.f / 32.f));
        float sn, cs; sincosf((float)s * freq, &sn, &cs);
        const float* kr = CKV + row * KVOc + KVPc;
        float rot = (j < 32) ? -kr[j + 32] : kr[j - 32];
        kval = kr[j] * cs + rot * sn;
    }
    size_t bh = (size_t)b * Hc + h;
    Kb[(bh * Sc + s) * DHc + d] = (bf16_t)kval;
    Vt[(bh * DHc + d) * Sc + s] = (bf16_t)KV[row * NKVc + h * 192 + 64 + d];
}

// ---------------------------------------------------------------- flash attention (causal)
__global__ __launch_bounds__(256) void k_mla_attn(const bf16_t* __restrict__ Qb,
                                                  const bf16_t* __restrict__ Kb,
                                                  const bf16_t* __restrict__ Vt,
                                                  bf16_t* __restrict__ AO) {
    __shared__ bf16_t Ps[8][16][40];
    int wave = threadIdx.x >> 5, lane = threadIdx.x & 31;
    int lh = lane >> 4, ll = lane & 15;
    int bh = blockIdx.x;                  // b*H + h
    int b = bh / Hc, h = bh % Hc;
    int q0 = blockIdx.y * 128 + wave * 16;

    const bf16_t* Qp = Qb + ((size_t)bh * Sc + q0) * DHc;
    const bf16_t* Kp = Kb + (size_t)bh * Sc * DHc;
    const bf16_t* Vp = Vt + (size_t)bh * DHc * Sc;

    v16bf qa[4];
#pragma unroll
    for (int kk = 0; kk < 4; ++kk)
        qa[kk] = load16(Qp + (size_t)ll * DHc + kk * 32 + lh * 16);

    v8f o[8];
#pragma unroll
    for (int dt = 0; dt < 8; ++dt) o[dt] = (v8f)0.f;
    float m_i[8], l_i[8];
#pragma unroll
    for (int r = 0; r < 8; ++r) { m_i[r] = NEG_BIG; l_i[r] = 0.f; }

    int qmax = q0 + 15;
    for (int t0 = 0; t0 <= qmax; t0 += 32) {
        v8f sc[2];
        sc[0] = (v8f)0.f; sc[1] = (v8f)0.f;
#pragma unroll
        for (int nt = 0; nt < 2; ++nt)
#pragma unroll
            for (int kk = 0; kk < 4; ++kk) {
                v16bf kf = load16(Kp + (size_t)(t0 + nt * 16 + ll) * DHc + kk * 32 + lh * 16);
                sc[nt] = wmma_bf16(qa[kk], kf, sc[nt]);
            }
        float alpha[8];
#pragma unroll
        for (int r = 0; r < 8; ++r) {
            int qrow = q0 + 8 * lh + r;
            float s0 = (t0 + ll > qrow) ? NEG_BIG : sc[0][r];
            float s1 = (t0 + 16 + ll > qrow) ? NEG_BIG : sc[1][r];
            float mx = fmaxf(s0, s1);
            mx = fmaxf(mx, __shfl_xor(mx, 1, 32));
            mx = fmaxf(mx, __shfl_xor(mx, 2, 32));
            mx = fmaxf(mx, __shfl_xor(mx, 4, 32));
            mx = fmaxf(mx, __shfl_xor(mx, 8, 32));
            float mnew = fmaxf(m_i[r], mx);
            float a = __expf(m_i[r] - mnew);
            m_i[r] = mnew;
            float p0 = __expf(s0 - mnew);
            float p1 = __expf(s1 - mnew);
            sc[0][r] = p0; sc[1][r] = p1;
            float rs = p0 + p1;
            rs += __shfl_xor(rs, 1, 32);
            rs += __shfl_xor(rs, 2, 32);
            rs += __shfl_xor(rs, 4, 32);
            rs += __shfl_xor(rs, 8, 32);
            l_i[r] = l_i[r] * a + rs;
            alpha[r] = a;
        }
#pragma unroll
        for (int dt = 0; dt < 8; ++dt)
#pragma unroll
            for (int r = 0; r < 8; ++r) o[dt][r] *= alpha[r];
        // P (accumulator layout) -> LDS -> A-fragment layout
#pragma unroll
        for (int nt = 0; nt < 2; ++nt)
#pragma unroll
            for (int r = 0; r < 8; ++r)
                Ps[wave][8 * lh + r][nt * 16 + ll] = (bf16_t)sc[nt][r];
        asm volatile("s_wait_dscnt 0" ::: "memory");   // wave-local DS ordering
        v16bf pa = load16(&Ps[wave][ll][lh * 16]);
#pragma unroll
        for (int dt = 0; dt < 8; ++dt) {
            v16bf vf = load16(Vp + (size_t)(dt * 16 + ll) * Sc + t0 + lh * 16);
            o[dt] = wmma_bf16(pa, vf, o[dt]);
        }
    }
#pragma unroll
    for (int r = 0; r < 8; ++r) {
        float inv = 1.f / l_i[r];
        int qrow = q0 + 8 * lh + r;
        bf16_t* dst = AO + ((size_t)b * Sc + qrow) * Dc + h * DHc;
#pragma unroll
        for (int dt = 0; dt < 8; ++dt)
            dst[dt * 16 + ll] = (bf16_t)(o[dt][r] * inv);
    }
}

// ---------------------------------------------------------------- launch
extern "C" void kernel_launch(void* const* d_in, const int* in_sizes, int n_in,
                              void* d_out, int out_size, void* d_ws, size_t ws_size,
                              hipStream_t stream) {
    const float* x      = (const float*)d_in[0];
    const float* W_dq   = (const float*)d_in[1];
    const float* W_uq   = (const float*)d_in[2];
    const float* q_ln_g = (const float*)d_in[3];
    const float* q_ln_b = (const float*)d_in[4];
    const float* W_dkv  = (const float*)d_in[5];
    const float* W_ukv  = (const float*)d_in[6];
    const float* kv_ln_g= (const float*)d_in[7];
    const float* kv_ln_b= (const float*)d_in[8];
    const float* W_o    = (const float*)d_in[9];

    float* out = (float*)d_out;                              // B*S*D
    float* ckv = out + (size_t)M4c * Dc;                     // B*S*KVO (2nd output)

    char* ws = (char*)d_ws;
    size_t off = 0;
    auto alloc = [&](size_t bytes) -> char* {
        off = (off + 255) & ~(size_t)255;
        char* p = ws + off;
        off += bytes;
        return p;
    };
    bf16_t* xb     = (bf16_t*)alloc((size_t)M4c * Dc * 2);
    bf16_t* Wdq_t  = (bf16_t*)alloc((size_t)QPc * Dc * 2);
    bf16_t* Wuq_t  = (bf16_t*)alloc((size_t)Dc * QPc * 2);
    bf16_t* Wdkv_t = (bf16_t*)alloc((size_t)KVOc * Dc * 2);
    bf16_t* Wukv_t = (bf16_t*)alloc((size_t)NKVc * KVPc * 2);
    bf16_t* Wo_b   = (bf16_t*)alloc((size_t)Dc * Dc * 2);
    float*  cq_raw = (float*)alloc((size_t)M4c * QPc * 4);
    bf16_t* cq_b   = (bf16_t*)alloc((size_t)M4c * QPc * 2);
    float*  Q_raw  = (float*)alloc((size_t)M4c * Dc * 4);
    bf16_t* Qb     = (bf16_t*)alloc((size_t)M4c * Dc * 2);
    bf16_t* kvl_b  = (bf16_t*)alloc((size_t)M4c * KVPc * 2);
    float*  KV_raw = (float*)alloc((size_t)M4c * NKVc * 4);
    bf16_t* Kb     = (bf16_t*)alloc((size_t)M4c * Dc * 2);   // B,H,S,128
    bf16_t* Vt_b   = (bf16_t*)alloc((size_t)M4c * Dc * 2);   // B,H,128,S
    bf16_t* AOb    = (bf16_t*)alloc((size_t)M4c * Dc * 2);
    (void)ws_size; (void)in_sizes; (void)n_in; (void)out_size;

    auto cdiv = [](size_t a, size_t b) { return (a + b - 1) / b; };

    // precision conversion + weight pre-transposition (N x K layout for NT GEMM)
    k_f32_to_bf16<<<cdiv((size_t)M4c * Dc, 256), 256, 0, stream>>>(x, xb, (size_t)M4c * Dc);
    k_transpose_bf16<<<dim3(cdiv(QPc, 32), cdiv(Dc, 32)), 256, 0, stream>>>(W_dq, Wdq_t, Dc, QPc);
    k_transpose_bf16<<<dim3(cdiv(Dc, 32), cdiv(QPc, 32)), 256, 0, stream>>>(W_uq, Wuq_t, QPc, Dc);
    k_transpose_bf16<<<dim3(cdiv(KVOc, 32), cdiv(Dc, 32)), 256, 0, stream>>>(W_dkv, Wdkv_t, Dc, KVOc);
    k_transpose_bf16<<<dim3(cdiv(NKVc, 32), cdiv(KVPc, 32)), 256, 0, stream>>>(W_ukv, Wukv_t, KVPc, NKVc);
    k_f32_to_bf16<<<cdiv((size_t)Dc * Dc, 256), 256, 0, stream>>>(W_o, Wo_b, (size_t)Dc * Dc);  // out@W_o^T: already N x K

    // cq = LN(x @ W_dq)
    k_gemm_nt<<<dim3(cdiv(QPc, BN), cdiv(M4c, BM)), 256, 0, stream>>>(xb, Wdq_t, cq_raw, M4c, QPc, Dc);
    k_layernorm<<<M4c, 256, 0, stream>>>(cq_raw, q_ln_g, q_ln_b, cq_b, QPc, QPc, QPc);

    // Q = cq @ W_uq ; rope + scale + (b,h,s,d) pack
    k_gemm_nt<<<dim3(cdiv(Dc, BN), cdiv(M4c, BM)), 256, 0, stream>>>(cq_b, Wuq_t, Q_raw, M4c, Dc, QPc);
    k_rope_q<<<cdiv((size_t)M4c * Dc, 256), 256, 0, stream>>>(Q_raw, Qb);

    // compressed_kv = x @ W_dkv  (2nd kernel output, fp32)
    k_gemm_nt<<<dim3(cdiv(KVOc, BN), cdiv(M4c, BM)), 256, 0, stream>>>(xb, Wdkv_t, ckv, M4c, KVOc, Dc);
    k_layernorm<<<M4c, 256, 0, stream>>>(ckv, kv_ln_g, kv_ln_b, kvl_b, KVPc, KVOc, KVPc);

    // KV = kv_lora @ W_ukv ; split into K(nope|rope) and V^T
    k_gemm_nt<<<dim3(cdiv(NKVc, BN), cdiv(M4c, BM)), 256, 0, stream>>>(kvl_b, Wukv_t, KV_raw, M4c, NKVc, KVPc);
    k_build_kv<<<cdiv((size_t)M4c * Dc, 256), 256, 0, stream>>>(KV_raw, ckv, Kb, Vt_b);

    // causal flash attention
    k_mla_attn<<<dim3(Bc * Hc, Sc / 128), 256, 0, stream>>>(Qb, Kb, Vt_b, AOb);

    // out = attn_out @ W_o^T
    k_gemm_nt<<<dim3(cdiv(Dc, BN), cdiv(M4c, BM)), 256, 0, stream>>>(AOb, Wo_b, out, M4c, Dc, Dc);
}
